// ICLAttentionExact_52493090292406
// MI455X (gfx1250) — compile-verified, matrix operand order checked by reading, not run on hardware
//
#include <hip/hip_runtime.h>
#include <cstdint>
#include <cstddef>

#define B_ 2
#define S_ 2048
#define E_ 1024
#define H_ 16

typedef __attribute__((ext_vector_type(16))) __bf16 v16bf;
typedef __attribute__((ext_vector_type(8)))  float  v8f;
typedef __attribute__((ext_vector_type(4)))  unsigned int uivec4;
typedef __attribute__((ext_vector_type(4)))  unsigned int u32x4;
typedef __attribute__((ext_vector_type(8)))  int          i32x8;
typedef __attribute__((ext_vector_type(4)))  int          i32x4;

#if defined(__has_builtin)
#  if __has_builtin(__builtin_amdgcn_tensor_load_to_lds)
#    define HAS_TDM 1
#  else
#    define HAS_TDM 0
#  endif
#else
#  define HAS_TDM 0
#endif

union FragBF { v16bf v; uivec4 u[2]; };

__device__ __forceinline__ unsigned short f2bf(float f) {
  unsigned u = __float_as_uint(f);
  unsigned r = (u + 0x7fffu + ((u >> 16) & 1u)) >> 16;   // round-to-nearest-even
  return (unsigned short)r;
}

// Load a 16x32 bf16 WMMA operand fragment from a row-major bf16 matrix with
// row stride `ld` (elements). Valid as the A operand (rows = M), and as the B
// operand when the source matrix is stored [N][K] row-major (rows = N).
// CDNA5 16-bit layout: lane L holds row (L&15); K elements are two contiguous
// 8-element runs at K offsets {lo, 16+lo}, lo = (L<16 ? 0 : 8).
__device__ __forceinline__ v16bf load_frag(const unsigned short* base, int ld, int lane) {
  int r  = lane & 15;
  int lo = (lane & 16) ? 8 : 0;
  const unsigned short* p = base + (size_t)r * ld + lo;
  FragBF f;
  f.u[0] = *reinterpret_cast<const uivec4*>(p);        // K = lo .. lo+7
  f.u[1] = *reinterpret_cast<const uivec4*>(p + 16);   // K = 16+lo .. 16+lo+7
  return f.v;
}

__device__ __forceinline__ v8f wmma_bf16(v16bf a, v16bf b, v8f c) {
  return __builtin_amdgcn_wmma_f32_16x16x32_bf16(false, a, false, b, (short)0, c, false, false);
}

#if HAS_TDM
// Issue one TDM 2D tile load: `rows` x E_ bf16 tile, row-major (stride E_),
// global -> LDS. Descriptor per CDNA5 D# (group0: count/lds/global/type=2;
// group1: data_size=2B, tensor_dim0=E_, tensor_dim1=rows, tile=E_ x rows,
// tensor_dim0_stride=E_). Groups 2/3 zero (tensor is 2-D). Wave-level op.
// This toolchain's builtin takes 6 args (g0, g1, g2, g3, g4, cpol).
__device__ __forceinline__ void tdm_load_2d(unsigned lds_addr,
                                            const unsigned short* gptr,
                                            int rows) {
  unsigned long long ga = (unsigned long long)(uintptr_t)gptr;
  u32x4 g0;
  g0[0] = 1u;                                     // count=1 valid descriptor
  g0[1] = lds_addr;                               // LDS byte address
  g0[2] = (unsigned)(ga & 0xFFFFFFFFu);           // global_addr[31:0]
  g0[3] = (unsigned)((ga >> 32) & 0x01FFFFFFu) | (2u << 30);  // addr[56:32] | type=2
  i32x8 g1;
  g1[0] = (1 << 16);                              // data_size=1 (2 bytes); no mask
  g1[1] = (int)((unsigned)E_ << 16);              // tensor_dim0[15:0] @ bits 63:48
  g1[2] = (int)(((unsigned)rows & 0xFFFFu) << 16);// dim0 hi=0 | tensor_dim1[15:0]
  g1[3] = (int)((unsigned)E_ << 16);              // dim1 hi=0 | tile_dim0=E_
  g1[4] = rows;                                   // tile_dim1 | tile_dim2=0
  g1[5] = E_;                                     // tensor_dim0_stride[31:0]
  g1[6] = 0;                                      // stride0 hi | stride1 lo
  g1[7] = 0;
  i32x4 z4 = {0, 0, 0, 0};
  i32x8 z8 = {0, 0, 0, 0, 0, 0, 0, 0};
  __builtin_amdgcn_tensor_load_to_lds(g0, g1, z4, z4, z8, 0);
}
#endif

// ---------------------------------------------------------------------------
// K0: qh/kh = bf16( rope( q * W_diag[h] ) ), stored [b,h,s,e] in workspace.
// ---------------------------------------------------------------------------
__global__ void icl_prep_kernel(const float* __restrict__ q, const float* __restrict__ k,
                                const float* __restrict__ W,
                                unsigned short* __restrict__ qh,
                                unsigned short* __restrict__ kh) {
  int s = blockIdx.x % S_;
  int h = (blockIdx.x / S_) % H_;
  int b = blockIdx.x / (S_ * H_);
  const float* qrow = q + (size_t)(b * S_ + s) * E_;
  const float* krow = k + (size_t)(b * S_ + s) * E_;
  const float* wrow = W + (size_t)h * E_;
  size_t dstoff = ((size_t)(b * H_ + h) * S_ + s) * E_;
  unsigned short* qdst = qh + dstoff;
  unsigned short* kdst = kh + dstoff;
  const float kLog2Base = 13.2877123795494f / (float)E_;   // log2(10000)/E
  #pragma unroll
  for (int p = 0; p < 4; ++p) {
    int j = threadIdx.x + p * 128;       // pair index 0..511
    int e = 2 * j;
    float inv = exp2f(-(float)e * kLog2Base);
    float ang = (float)s * inv;
    float sn, cs;
    __sincosf(ang, &sn, &cs);
    float w0 = wrow[e], w1 = wrow[e + 1];
    float a0 = qrow[e] * w0, a1 = qrow[e + 1] * w1;
    qdst[e]     = f2bf(a0 * cs - a1 * sn);
    qdst[e + 1] = f2bf(a1 * cs + a0 * sn);
    float c0 = krow[e] * w0, c1 = krow[e + 1] * w1;
    kdst[e]     = f2bf(c0 * cs - c1 * sn);
    kdst[e + 1] = f2bf(c1 * cs + c0 * sn);
  }
}

// ---------------------------------------------------------------------------
// K1: per-row softmax stats (m = rowmax, l = sum exp) over k <= q.
// Grid: (b,h,q-tile16). Query tile staged via TDM; 8 waves split k-tiles;
// online softmax in registers, shuffles for cross-lane, LDS for cross-wave.
// ---------------------------------------------------------------------------
__global__ void icl_stats_kernel(const unsigned short* __restrict__ qh,
                                 const unsigned short* __restrict__ kh,
                                 float* __restrict__ mws, float* __restrict__ lws) {
  __shared__ unsigned short qtile[16 * E_];   // 32 KB: this tile's 16 query rows
  __shared__ float redm[8 * 16];
  __shared__ float redl[8 * 16];

  int qt = blockIdx.x % (S_ / 16);
  int h  = (blockIdx.x / (S_ / 16)) % H_;
  int b  = blockIdx.x / ((S_ / 16) * H_);
  int bh = b * H_ + h;
  int q0 = qt * 16;
  int tid = threadIdx.x;
  int wave = tid >> 5, lane = tid & 31;

#if HAS_TDM
  if (tid < 32) {
    tdm_load_2d((unsigned)(uintptr_t)qtile, qh + ((size_t)bh * S_ + q0) * E_, 16);
    __builtin_amdgcn_s_wait_tensorcnt(0);
  }
#else
  {
    const uivec4* src = reinterpret_cast<const uivec4*>(qh + ((size_t)bh * S_ + q0) * E_);
    uivec4* dst = reinterpret_cast<uivec4*>(qtile);
    for (int i = tid; i < 16 * E_ / 8; i += 256) dst[i] = src[i];
  }
#endif
  __syncthreads();

  const float NEG_INF = -__builtin_inff();
  float m[8], l[8];
  #pragma unroll
  for (int r = 0; r < 8; ++r) { m[r] = NEG_INF; l[r] = 0.f; }

  int N  = lane & 15;
  int hi = (lane & 16) ? 8 : 0;

  for (int kt = wave; kt <= qt; kt += 8) {
    v8f sc = {};
    const unsigned short* kbase = kh + ((size_t)bh * S_ + kt * 16) * E_;
    #pragma unroll 4
    for (int ec = 0; ec < E_ / 32; ++ec) {
      v16bf a  = load_frag(qtile + ec * 32, E_, lane);
      v16bf bb = load_frag(kbase + ec * 32, E_, lane);
      sc = wmma_bf16(a, bb, sc);
    }
    int kg = kt * 16 + N;
    #pragma unroll
    for (int r = 0; r < 8; ++r) {
      int qg = q0 + hi + r;
      float val = (kg > qg) ? NEG_INF : sc[r];   // causal mask (diag kept for stats)
      float tm = val;
      tm = fmaxf(tm, __shfl_xor(tm, 1));
      tm = fmaxf(tm, __shfl_xor(tm, 2));
      tm = fmaxf(tm, __shfl_xor(tm, 4));
      tm = fmaxf(tm, __shfl_xor(tm, 8));
      float nm = fmaxf(m[r], tm);
      float p = __expf(val - nm);
      p += __shfl_xor(p, 1);
      p += __shfl_xor(p, 2);
      p += __shfl_xor(p, 4);
      p += __shfl_xor(p, 8);
      l[r] = l[r] * __expf(m[r] - nm) + p;
      m[r] = nm;
    }
  }
  if ((lane & 15) == 0) {
    #pragma unroll
    for (int r = 0; r < 8; ++r) {
      redm[wave * 16 + hi + r] = m[r];
      redl[wave * 16 + hi + r] = l[r];
    }
  }
  __syncthreads();
  if (tid < 16) {
    float M = NEG_INF;
    for (int w = 0; w < 8; ++w) M = fmaxf(M, redm[w * 16 + tid]);
    float L = 0.f;
    for (int w = 0; w < 8; ++w) L += redl[w * 16 + tid] * __expf(redm[w * 16 + tid] - M);
    mws[(size_t)bh * S_ + q0 + tid] = M;
    lws[(size_t)bh * S_ + q0 + tid] = L;
  }
}

// ---------------------------------------------------------------------------
// K2: out[b, k0+kk, :] = (1/(k+1)) * sum_h lr_h * sum_{q>k} P[q,k] v[q,:]
// Grid: (b, k-tile16). Per-head Q/K tiles DMA'd into double-buffered LDS by
// the TDM (prefetch of head h+1 overlaps WMMA work of head h). Head-sum
// accumulators live in registers; cross-wave score reduction through LDS
// (deterministic, no float atomics).
// ---------------------------------------------------------------------------
__global__ void icl_out_kernel(const unsigned short* __restrict__ qh,
                               const unsigned short* __restrict__ kh,
                               const float* __restrict__ vin,
                               const float* __restrict__ lr_values,
                               const float* __restrict__ mws,
                               const float* __restrict__ lws,
                               float* __restrict__ out) {
  extern __shared__ char smem[];
  float* sp   = reinterpret_cast<float*>(smem);                 // [8][32][16] partial scores, 16 KB
  float* mrow = sp + 8 * 32 * 16;                               // [32]
  float* linv = mrow + 32;                                      // [32]
  unsigned short* vt  = reinterpret_cast<unsigned short*>(linv + 32); // [E][32] V^T bf16, 64 KB
  unsigned short* pt  = vt + (size_t)E_ * 32;                   // [16][32] P^T bf16, 1 KB
  unsigned short* kb0 = pt + 16 * 32;                           // K tile buf 0, 32 KB
  unsigned short* kb1 = kb0 + 16 * E_;                          // K tile buf 1, 32 KB
  unsigned short* qb0 = kb1 + 16 * E_;                          // Q tile buf 0, 64 KB
  unsigned short* qb1 = qb0 + 32 * E_;                          // Q tile buf 1, 64 KB

  int kt = blockIdx.x % (S_ / 16);
  int b  = blockIdx.x / (S_ / 16);
  int k0 = kt * 16;
  int tid = threadIdx.x;
  int wave = tid >> 5, lane = tid & 31;
  int N  = lane & 15;
  int hi = (lane & 16) ? 8 : 0;

  unsigned short* kbufs[2] = { kb0, kb1 };
  unsigned short* qbufs[2] = { qb0, qb1 };

  v8f acc[8];
  #pragma unroll
  for (int nt = 0; nt < 8; ++nt) { v8f z = {}; acc[nt] = z; }

  int qs0 = k0 & ~31;
  for (int qs = qs0; qs < S_; qs += 32) {
#if HAS_TDM
    // Kick off DMA of head 0's K/Q tiles; overlaps with V^T staging below.
    if (tid < 32) {
      int bh0 = b * H_;
      tdm_load_2d((unsigned)(uintptr_t)kbufs[0], kh + ((size_t)bh0 * S_ + k0) * E_, 16);
      tdm_load_2d((unsigned)(uintptr_t)qbufs[0], qh + ((size_t)bh0 * S_ + qs) * E_, 32);
    }
#endif
    // Stage V^T (bf16): vt[e][qq] = v[b, qs+qq, e]; shared across the h loop.
    for (int i = tid; i < 32 * E_; i += 256) {
      int qq = i >> 10;                 // E_ == 1024
      int e  = i & (E_ - 1);
      float x = vin[(size_t)(b * S_ + qs + qq) * E_ + e];
      vt[e * 32 + qq] = f2bf(x);
    }
    __syncthreads();

    for (int h = 0; h < H_; ++h) {
      int bh  = b * H_ + h;
      int cur = h & 1;
#if !HAS_TDM
      {  // fallback: cooperative copy of this head's K/Q tiles
        const uivec4* ks = reinterpret_cast<const uivec4*>(kh + ((size_t)bh * S_ + k0) * E_);
        const uivec4* qsrc = reinterpret_cast<const uivec4*>(qh + ((size_t)bh * S_ + qs) * E_);
        uivec4* kd = reinterpret_cast<uivec4*>(kbufs[cur]);
        uivec4* qd = reinterpret_cast<uivec4*>(qbufs[cur]);
        for (int i = tid; i < 16 * E_ / 8; i += 256) kd[i] = ks[i];
        for (int i = tid; i < 32 * E_ / 8; i += 256) qd[i] = qsrc[i];
      }
#endif
      if (tid < 32) {
#if HAS_TDM
        __builtin_amdgcn_s_wait_tensorcnt(0);   // this head's tiles are in LDS
#endif
        mrow[tid] = mws[(size_t)bh * S_ + qs + tid];
        linv[tid] = 1.0f / lws[(size_t)bh * S_ + qs + tid];
      }
      __syncthreads();                          // publish tiles + stats
#if HAS_TDM
      if (h + 1 < H_ && tid < 32) {             // prefetch next head's tiles
        int bhn = bh + 1;
        tdm_load_2d((unsigned)(uintptr_t)kbufs[1 - cur],
                    kh + ((size_t)bhn * S_ + k0) * E_, 16);
        tdm_load_2d((unsigned)(uintptr_t)qbufs[1 - cur],
                    qh + ((size_t)bhn * S_ + qs) * E_, 32);
      }
#endif
      // Scores: each wave reduces its 4 e-chunks (128 of E) -> partial frags.
      const unsigned short* kbase = kbufs[cur];
      #pragma unroll
      for (int half = 0; half < 2; ++half) {
        const unsigned short* qbase = qbufs[cur] + half * 16 * E_;
        v8f sc = {};
        #pragma unroll
        for (int j = 0; j < 4; ++j) {
          int ec = wave * 4 + j;
          v16bf a  = load_frag(qbase + ec * 32, E_, lane);
          v16bf bb = load_frag(kbase + ec * 32, E_, lane);
          sc = wmma_bf16(a, bb, sc);
        }
        #pragma unroll
        for (int r = 0; r < 8; ++r)
          sp[(wave * 32 + half * 16 + hi + r) * 16 + N] = sc[r];
      }
      __syncthreads();

      // Build P^T bf16 (exp, 1/l, lr_h, strict causal + zero diagonal).
      float lr = lr_values[h];
      for (int i = tid; i < 512; i += 256) {
        int qq = i >> 4;
        int kk = i & 15;
        float s = 0.f;
        #pragma unroll
        for (int w = 0; w < 8; ++w) s += sp[(w * 32 + qq) * 16 + kk];
        float p = 0.f;
        int qg = qs + qq, kg = k0 + kk;
        if (kg < qg) p = __expf(s - mrow[qq]) * linv[qq] * lr;
        pt[kk * 32 + qq] = f2bf(p);
      }
      __syncthreads();

      // acc[16k x 16e] += P^T (16k x 32q) * V (32q x 16e), one WMMA per n-tile.
      v16bf a = load_frag(pt, 32, lane);
      #pragma unroll
      for (int nt = 0; nt < 8; ++nt) {
        int e0 = wave * 128 + nt * 16;
        v16bf bb = load_frag(vt + (size_t)e0 * 32, 32, lane);
        acc[nt] = wmma_bf16(a, bb, acc[nt]);
      }
    }
    __syncthreads();   // protect vt/mrow/tile buffers before restaging
  }

  // Epilogue: every output element written exactly once, scaled by 1/(s+1).
  #pragma unroll
  for (int nt = 0; nt < 8; ++nt) {
    int e = wave * 128 + nt * 16 + N;
    #pragma unroll
    for (int r = 0; r < 8; ++r) {
      int sg = k0 + hi + r;
      out[(size_t)(b * S_ + sg) * E_ + e] = acc[nt][r] * (1.0f / (float)(sg + 1));
    }
  }
}

extern "C" void kernel_launch(void* const* d_in, const int* in_sizes, int n_in,
                              void* d_out, int out_size, void* d_ws, size_t ws_size,
                              hipStream_t stream) {
  (void)in_sizes; (void)n_in; (void)out_size; (void)ws_size;
  const float* q  = (const float*)d_in[0];
  const float* k  = (const float*)d_in[1];
  const float* v  = (const float*)d_in[2];
  const float* W  = (const float*)d_in[3];
  const float* lr = (const float*)d_in[4];
  float* out = (float*)d_out;

  const size_t nElem = (size_t)B_ * H_ * S_ * E_;
  unsigned short* qh = (unsigned short*)d_ws;
  unsigned short* kh = qh + nElem;
  float* mws = (float*)(kh + nElem);
  float* lws = mws + (size_t)B_ * H_ * S_;

  icl_prep_kernel<<<B_ * H_ * S_, 128, 0, stream>>>(q, k, W, qh, kh);
  icl_stats_kernel<<<B_ * H_ * (S_ / 16), 256, 0, stream>>>(qh, kh, mws, lws);

  size_t dynLDS = (size_t)8 * 32 * 16 * sizeof(float)        // sp
                + 64 * sizeof(float)                         // mrow + linv
                + (size_t)E_ * 32 * sizeof(unsigned short)   // vt
                + (size_t)16 * 32 * sizeof(unsigned short)   // pt
                + 2 * (size_t)16 * E_ * sizeof(unsigned short)   // K tile x2
                + 2 * (size_t)32 * E_ * sizeof(unsigned short);  // Q tile x2
  (void)hipFuncSetAttribute((const void*)icl_out_kernel,
                            hipFuncAttributeMaxDynamicSharedMemorySize, (int)dynLDS);
  icl_out_kernel<<<B_ * (S_ / 16), 256, dynLDS, stream>>>(qh, kh, v, lr, mws, lws, out);
}
